// TernaryLinearTriton_37838661877796
// MI455X (gfx1250) — 8-bit matrix operand order repaired and checked by reading, compiled, not run on hardware
//
#include <hip/hip_runtime.h>


typedef int    v16i __attribute__((ext_vector_type(16)));
typedef float  v8f  __attribute__((ext_vector_type(8)));
typedef int    i2   __attribute__((ext_vector_type(2)));
typedef int    i4   __attribute__((ext_vector_type(4)));
typedef float  f4   __attribute__((ext_vector_type(4)));
typedef unsigned char uc4 __attribute__((ext_vector_type(4)));

#define DELTA    0.05f
#define FP8_ONE  ((unsigned char)0x38)
#define FP8_NEG1 ((unsigned char)0xB8)

static __device__ __forceinline__ unsigned char tern_fp8(float v) {
    return (v > DELTA) ? FP8_ONE : ((v < -DELTA) ? FP8_NEG1 : (unsigned char)0);
}

__global__ __launch_bounds__(256) void quant_x_kernel(const float* __restrict__ x,
                                                      unsigned char* __restrict__ aq,
                                                      long long n4) {
    long long i = (long long)blockIdx.x * blockDim.x + threadIdx.x;
    if (i >= n4) return;
    f4 v = __builtin_nontemporal_load((const f4*)x + i);
    uc4 o;
    o.x = tern_fp8(v.x); o.y = tern_fp8(v.y); o.z = tern_fp8(v.z); o.w = tern_fp8(v.w);
    *((uc4*)aq + i) = o;
}

__global__ __launch_bounds__(256) void quant_w_kernel(const float* __restrict__ w,
                                                      unsigned char* __restrict__ wp,
                                                      unsigned char* __restrict__ wn,
                                                      long long n4) {
    long long i = (long long)blockIdx.x * blockDim.x + threadIdx.x;
    if (i >= n4) return;
    f4 v = __builtin_nontemporal_load((const f4*)w + i);
    uc4 p, q;
    p.x = (v.x >  DELTA) ? FP8_ONE : 0;  q.x = (v.x < -DELTA) ? FP8_ONE : 0;
    p.y = (v.y >  DELTA) ? FP8_ONE : 0;  q.y = (v.y < -DELTA) ? FP8_ONE : 0;
    p.z = (v.z >  DELTA) ? FP8_ONE : 0;  q.z = (v.z < -DELTA) ? FP8_ONE : 0;
    p.w = (v.w >  DELTA) ? FP8_ONE : 0;  q.w = (v.w < -DELTA) ? FP8_ONE : 0;
    *((uc4*)wp + i) = p;
    *((uc4*)wn + i) = q;
}

__global__ __launch_bounds__(256, 1) void ternary_wmma_gemm(
    const unsigned char* __restrict__ Aq,
    const unsigned char* __restrict__ Wp,
    const unsigned char* __restrict__ Wn,
    const float* __restrict__ alpha_p,
    const float* __restrict__ alpha_n,
    const float* __restrict__ bias,
    float*       __restrict__ out,
    int M, int N, int K)
{
    const int lane = threadIdx.x & 31;
    const int wave = threadIdx.x >> 5;
    const int wm   = wave & 1;
    const int wn   = wave >> 1;
    const int half = lane >> 4;
    const int lrow = lane & 15;

    const int waveM = blockIdx.y * 128 + wm * 64;
    const int waveN = blockIdx.x * 128 + wn * 32;

    v8f accP[4][2] = {};
    v8f accN[4][2] = {};

    const unsigned char* arow[4];
#pragma unroll
    for (int mi = 0; mi < 4; ++mi)
        arow[mi] = Aq + (size_t)(waveM + mi * 16 + lrow) * K + half * 16;

    const unsigned char* bprow[2];
    const unsigned char* bnrow[2];
#pragma unroll
    for (int ni = 0; ni < 2; ++ni) {
        size_t off = (size_t)(waveN + ni * 16 + lrow) * K + half * 16;
        bprow[ni] = Wp + off;
        bnrow[ni] = Wn + off;
    }

    for (int k0 = 0; k0 < K; k0 += 128) {
        v16i a[4];
#pragma unroll
        for (int mi = 0; mi < 4; ++mi) {
            const unsigned char* p = arow[mi] + k0;
#pragma unroll
            for (int c = 0; c < 8; ++c) {
                i2 t = *(const i2*)(p + (c >> 1) * 32 + (c & 1) * 8);
                a[mi][2 * c]     = t.x;
                a[mi][2 * c + 1] = t.y;
            }
        }
#pragma unroll
        for (int ni = 0; ni < 2; ++ni) {
            v16i bp, bn;
#pragma unroll
            for (int c = 0; c < 4; ++c) {
                i4 u = *(const i4*)(bprow[ni] + k0 + c * 32);
                bp[4 * c] = u.x; bp[4 * c + 1] = u.y; bp[4 * c + 2] = u.z; bp[4 * c + 3] = u.w;
            }
#pragma unroll
            for (int c = 0; c < 4; ++c) {
                i4 u = *(const i4*)(bnrow[ni] + k0 + c * 32);
                bn[4 * c] = u.x; bn[4 * c + 1] = u.y; bn[4 * c + 2] = u.z; bn[4 * c + 3] = u.w;
            }
#pragma unroll
            for (int mi = 0; mi < 4; ++mi) {
                accP[mi][ni] = __builtin_amdgcn_wmma_f32_16x16x128_fp8_fp8(
                    a[mi], bp, (short)0, accP[mi][ni], false, false);
                accN[mi][ni] = __builtin_amdgcn_wmma_f32_16x16x128_fp8_fp8(
                    a[mi], bn, (short)0, accN[mi][ni], false, false);
            }
        }
    }

#pragma unroll
    for (int ni = 0; ni < 2; ++ni) {
        const int n  = waveN + ni * 16 + lrow;
        const float ap = alpha_p[n];
        const float an = alpha_n[n];
        const float bb = bias[n];
#pragma unroll
        for (int mi = 0; mi < 4; ++mi) {
            const int mbase = waveM + mi * 16 + half * 8;
#pragma unroll
            for (int r = 0; r < 8; ++r) {
                float v = ap * accP[mi][ni][r] - an * accN[mi][ni][r] + bb;
                __builtin_nontemporal_store(v, &out[(size_t)(mbase + r) * N + n]);
            }
        }
    }
}

extern "C" void kernel_launch(void* const* d_in, const int* in_sizes, int n_in,
                              void* d_out, int out_size, void* d_ws, size_t ws_size,
                              hipStream_t stream) {
    const float* x        = (const float*)d_in[0];
    const float* w_latent = (const float*)d_in[1];
    const float* alpha_p  = (const float*)d_in[2];
    const float* alpha_n  = (const float*)d_in[3];
    const float* bias     = (const float*)d_in[4];
    float* out = (float*)d_out;

    const int N = in_sizes[2];
    const int K = in_sizes[1] / N;
    const int M = in_sizes[0] / K;

    unsigned char* Aq = (unsigned char*)d_ws;
    unsigned char* Wp = Aq + (size_t)M * K;
    unsigned char* Wn = Wp + (size_t)N * K;

    {
        long long n4 = (long long)M * K / 4;
        quant_x_kernel<<<dim3((unsigned)((n4 + 255) / 256)), dim3(256), 0, stream>>>(x, Aq, n4);
    }
    {
        long long n4 = (long long)N * K / 4;
        quant_w_kernel<<<dim3((unsigned)((n4 + 255) / 256)), dim3(256), 0, stream>>>(w_latent, Wp, Wn, n4);
    }

    dim3 grid(N / 128, M / 128);
    ternary_wmma_gemm<<<grid, dim3(256), 0, stream>>>(
        Aq, Wp, Wn, alpha_p, alpha_n, bias, out, M, N, K);
}
